// Router_8564164788845
// MI455X (gfx1250) — matrix operand order repaired and checked by reading, not compiled, not run on hardware
//
#include <hip/hip_runtime.h>
#include <hip/hip_bf16.h>
#include <stdint.h>

// ---------------------------------------------------------------------------
// MoE router for MI455X (gfx1250): logits = x @ W^T + bias, softmax, top-2,
// renormalize.  HBM-bound on streaming x (256 MB @ 23.3 TB/s ~ 11.5us):
// bf16 WMMA (v_wmma_f32_16x16x32_bf16) makes compute free; the tiny gate
// matrix is staged in LDS as bf16; x is streamed fp32 -> bf16 in registers.
//
// Round-2 fix: the scheduler kept folding all B fragments into one register
// set, re-serializing ds_load -> s_wait_dscnt 0 -> wmma.  Now B is software-
// pipelined one k-step ahead and raw A two k-steps ahead (rotating register
// sets that are simultaneously live), with sched_barrier(0) fences so the
// scheduler cannot sink the prefetch loads below the WMMA group.
// ---------------------------------------------------------------------------

typedef __attribute__((ext_vector_type(16))) __bf16 bf16x16;
typedef __attribute__((ext_vector_type(8)))  float  f32x8;

#define D_MODEL      2048
#define NUM_EXPERTS  64
#define K_CHUNK      512
#define N_CHUNKS     (D_MODEL / K_CHUNK)     // 4
#define K_STEPS      (K_CHUNK / 32)          // 16 wmma k-steps per chunk
#define ROWS_PER_WG  128
#define THREADS      256                     // 8 waves (wave32)
#define WAVES        8
#define LOGIT_PITCH  68                      // 64 + 4 pad -> conflict-free LDS

#define LDS_GATE_BYTES   (NUM_EXPERTS * K_CHUNK * 2)          // 65536 (bf16)
#define LDS_LOGIT_BYTES  (WAVES * 16 * LOGIT_PITCH * 4)       // 34816
#define LDS_TOTAL        (LDS_GATE_BYTES + LDS_LOGIT_BYTES)   // 100352

// Pack two fp32 into packed bf16 pair (round-half-up) with a single v_perm.
__device__ __forceinline__ uint32_t pack_bf16(float a, float b) {
    uint32_t au = __builtin_bit_cast(uint32_t, a) + 0x8000u;
    uint32_t bu = __builtin_bit_cast(uint32_t, b) + 0x8000u;
    return __builtin_amdgcn_perm(bu, au, 0x07060302u);
}

union AFrag { uint32_t u[8]; bf16x16 v; };
union BFrag { uint4 q[2];    bf16x16 v; };
struct RawA { float4 a0, a1, a2, a3; };   // 16 fp32 (one A fragment, raw)

__device__ __forceinline__ RawA load_rawA(const float* p) {
    RawA r;
    r.a0 = *(const float4*)(p);
    r.a1 = *(const float4*)(p + 4);
    r.a2 = *(const float4*)(p + 16);
    r.a3 = *(const float4*)(p + 20);
    return r;
}

__device__ __forceinline__ AFrag convA(const RawA& r) {
    AFrag A;
    A.u[0] = pack_bf16(r.a0.x, r.a0.y);
    A.u[1] = pack_bf16(r.a0.z, r.a0.w);
    A.u[2] = pack_bf16(r.a1.x, r.a1.y);
    A.u[3] = pack_bf16(r.a1.z, r.a1.w);
    A.u[4] = pack_bf16(r.a2.x, r.a2.y);
    A.u[5] = pack_bf16(r.a2.z, r.a2.w);
    A.u[6] = pack_bf16(r.a3.x, r.a3.y);
    A.u[7] = pack_bf16(r.a3.z, r.a3.w);
    return A;
}

__device__ __forceinline__ void load_B(const uint32_t* bp, BFrag B[4]) {
    B[0].q[0] = ((const uint4*)(bp                      ))[0];
    B[0].q[1] = ((const uint4*)(bp                      ))[2];
    B[1].q[0] = ((const uint4*)(bp + 16 * (K_CHUNK / 2) ))[0];
    B[1].q[1] = ((const uint4*)(bp + 16 * (K_CHUNK / 2) ))[2];
    B[2].q[0] = ((const uint4*)(bp + 32 * (K_CHUNK / 2) ))[0];
    B[2].q[1] = ((const uint4*)(bp + 32 * (K_CHUNK / 2) ))[2];
    B[3].q[0] = ((const uint4*)(bp + 48 * (K_CHUNK / 2) ))[0];
    B[3].q[1] = ((const uint4*)(bp + 48 * (K_CHUNK / 2) ))[2];
}

__global__ __launch_bounds__(THREADS)
void router_wmma_kernel(const float* __restrict__ x,
                        const float* __restrict__ gate_w,
                        const float* __restrict__ expert_bias,
                        float* __restrict__ out_w,
                        int*   __restrict__ out_e,
                        int n_rows) {
    extern __shared__ char smem[];
    uint32_t* lds_gate   = (uint32_t*)smem;                    // [64][512] bf16
    float*    lds_logits = (float*)(smem + LDS_GATE_BYTES);    // [8][16][68] f32

    const int tid    = threadIdx.x;
    const int wave   = tid >> 5;
    const int lane   = tid & 31;
    const int lane16 = lane & 15;
    const int hi     = lane >> 4;            // 0 for lanes 0-15, 1 for 16-31

    const int m_base = blockIdx.x * ROWS_PER_WG + wave * 16;
    int a_row = m_base + lane16;
    if (a_row >= n_rows) a_row = n_rows - 1;         // clamp (never stored)
    const float* ap = x + (size_t)a_row * D_MODEL;

    // A fragment k-pattern (16-bit A 16x32): lanes<16 take K {0..7,16..23},
    // lanes>=16 take K {8..15,24..31}.
    const int khalfA = hi * 8;
    // B fragment (32x16): lane = column, same K pattern as A: lanes<16 hold
    // K {0..7,16..23}, lanes>=16 hold K {8..15,24..31}.
    const int khalfB = hi * 8;

    const uint32_t* bbase = lds_gate + (size_t)lane16 * (K_CHUNK / 2)
                                     + (khalfB >> 1);

    f32x8 acc[4] = {f32x8{}, f32x8{}, f32x8{}, f32x8{}};

    for (int c = 0; c < N_CHUNKS; ++c) {
        __syncthreads();   // previous chunk fully consumed before overwrite
        // ---- stage gate_w[:, c*512 : (c+1)*512] as bf16 into LDS ----
        const float* gw = gate_w + c * K_CHUNK;
        for (int i = tid * 4; i < NUM_EXPERTS * K_CHUNK; i += THREADS * 4) {
            const int e  = i >> 9;                 // / K_CHUNK
            const int kl = i & (K_CHUNK - 1);
            const float4 g = *(const float4*)(gw + (size_t)e * D_MODEL + kl);
            uint2 p;
            p.x = pack_bf16(g.x, g.y);
            p.y = pack_bf16(g.z, g.w);
            *(uint2*)(lds_gate + (i >> 1)) = p;    // ds_store_b64
        }
        __syncthreads();

        const float* abase = ap + c * K_CHUNK + khalfA;

        // ---- pipeline prologue: A raw two steps deep, B one step deep ----
        RawA raw0 = load_rawA(abase);              // ks = 0
        RawA raw1 = load_rawA(abase + 32);         // ks = 1
        BFrag Bc[4];
        load_B(bbase, Bc);                         // ks = 0

        #pragma unroll 4
        for (int ks = 0; ks < K_STEPS; ++ks) {
            // ---- issue next-step loads BEFORE consuming current ----
            BFrag Bn[4];
            load_B(bbase + (((ks + 1) & 15) * 32 >> 1), Bn);
            RawA rawn = load_rawA(abase + ((ks + 2) & 15) * 32);
            __builtin_prefetch(abase + ks * 32 + K_CHUNK, 0, 3);  // stream x

            AFrag A = convA(raw0);

            // Fence: loads above may not sink below the WMMA group.
            __builtin_amdgcn_sched_barrier(0);
            acc[0] = __builtin_amdgcn_wmma_f32_16x16x32_bf16(
                false, A.v, false, Bc[0].v, (short)0, acc[0], false, false);
            acc[1] = __builtin_amdgcn_wmma_f32_16x16x32_bf16(
                false, A.v, false, Bc[1].v, (short)0, acc[1], false, false);
            acc[2] = __builtin_amdgcn_wmma_f32_16x16x32_bf16(
                false, A.v, false, Bc[2].v, (short)0, acc[2], false, false);
            acc[3] = __builtin_amdgcn_wmma_f32_16x16x32_bf16(
                false, A.v, false, Bc[3].v, (short)0, acc[3], false, false);
            __builtin_amdgcn_sched_barrier(0);

            // ---- rotate pipeline registers ----
            raw0 = raw1;
            raw1 = rawn;
            Bc[0] = Bn[0]; Bc[1] = Bn[1]; Bc[2] = Bn[2]; Bc[3] = Bn[3];
        }
    }

    // ---- stage this wave's 16x64 logits to padded LDS slab ----
    float* Lw = lds_logits + (size_t)wave * 16 * LOGIT_PITCH;
    #pragma unroll
    for (int t = 0; t < 4; ++t) {
        const int col = t * 16 + lane16;
        #pragma unroll
        for (int j = 0; j < 8; ++j) {
            const int row = hi * 8 + j;        // C layout: lanes>=16 -> rows 8..15
            Lw[row * LOGIT_PITCH + col] = acc[t][j];
        }
    }
    __syncthreads();   // cross-lane exchange within/between waves

    // ---- softmax + top-2 + renormalize: one lane per row ----
    if (lane < 16) {
        const int    row = lane16;
        const float* L   = Lw + row * LOGIT_PITCH;

        float v1 = -INFINITY, v2 = -INFINITY;
        int   i1 = 0,         i2 = 0;
        for (int j = 0; j < NUM_EXPERTS; ++j) {
            const float v = L[j] + expert_bias[j];
            if (v > v1)      { v2 = v1; i2 = i1; v1 = v; i1 = j; }
            else if (v > v2) { v2 = v;  i2 = j; }
        }
        const float mx = v1;                   // max logit == top-1
        float Z = 0.f;
        for (int j = 0; j < NUM_EXPERTS; ++j)
            Z += __expf(L[j] + expert_bias[j] - mx);

        const float e1 = 1.0f;                 // exp(v1 - mx)
        const float e2 = __expf(v2 - mx);
        // w_k = p_k / (p1 + p2 + eps), p_k = e_k / Z => e_k / (e1+e2+eps*Z)
        const float denom = e1 + e2 + 1e-8f * Z;
        const float w1 = e1 / denom;
        const float w2 = e2 / denom;

        const int g = m_base + row;
        if (g < n_rows) {
            ((float2*)out_w)[g] = make_float2(w1, w2);
            ((int2*)out_e)[g]   = make_int2(i1, i2);
        }
    }
}

extern "C" void kernel_launch(void* const* d_in, const int* in_sizes, int n_in,
                              void* d_out, int out_size, void* d_ws, size_t ws_size,
                              hipStream_t stream) {
    const float* x    = (const float*)d_in[0];
    const float* gw   = (const float*)d_in[1];
    const float* bias = (const float*)d_in[2];

    const int n_rows = in_sizes[0] / D_MODEL;     // 32768

    // d_out = [routing_weights (N,2) f32 | selected_experts (N,2) i32]
    float* out_w = (float*)d_out;
    int*   out_e = (int*)((float*)d_out + 2 * (size_t)n_rows);

    (void)hipFuncSetAttribute((const void*)router_wmma_kernel,
                              hipFuncAttributeMaxDynamicSharedMemorySize,
                              LDS_TOTAL);

    const int grid = (n_rows + ROWS_PER_WG - 1) / ROWS_PER_WG;
    router_wmma_kernel<<<grid, THREADS, LDS_TOTAL, stream>>>(
        x, gw, bias, out_w, out_e, n_rows);
}
